// PolyLoss_4071628996806
// MI455X (gfx1250) — compile-verified, matrix operand order checked by reading, not run on hardware
//
#include <hip/hip_runtime.h>
#include <hip/hip_bf16.h>

typedef __attribute__((ext_vector_type(16))) _Float16 v16h;
typedef __attribute__((ext_vector_type(8)))  float    v8f;

#define B_   4
#define K_   128
#define C_   32
#define H_   128
#define W_   128
#define V_   16
#define NPOLY (B_ * K_)   // 512

// One block (256 threads = 8 wave32) per (b,k) polygon pair.
// Rasterizes pred & target polygons with even-odd parity; per-pixel edge
// counts are computed with V_WMMA_F32_16X16X32_F16 (A = indicator matrix,
// B = all-ones so D[m][*] = sum_k A[m,k] regardless of B register layout).
__global__ void __launch_bounds__(256) poly_iou_kernel(
    const float* __restrict__ output,
    const int*   __restrict__ mask,
    const int*   __restrict__ ind,
    const float* __restrict__ target,
    float*       __restrict__ ws_iou,
    float*       __restrict__ ws_reg)
{
    __shared__ float xsP[V_], ysP[V_], xsG[V_], ysG[V_];
    __shared__ float predSh[C_];
    __shared__ float redF[256];
    __shared__ int   redI[3 * 256];

    const int bk   = blockIdx.x;
    const int b    = bk / K_;
    const int tid  = threadIdx.x;
    const int lane = tid & 31;

    const float m = (float)mask[bk];

    // ---- gather pred + masked L1 partial ----
    float absdiff = 0.0f;
    if (tid < C_) {
        const int p  = ind[bk];
        const float pv = output[(b * C_ + tid) * (H_ * W_) + p];
        const float tv = target[bk * C_ + tid];
        predSh[tid] = pv;
        absdiff = fabsf(pv * m - tv * m);
    }
    redF[tid] = absdiff;
    __syncthreads();

    // ---- build polygon vertices (x = ch 0::2 + H/4, y = ch 1::2 + W/4) ----
    if (tid < V_) {
        xsP[tid] = predSh[2 * tid]     + (float)(H_ / 4);
        ysP[tid] = predSh[2 * tid + 1] + (float)(W_ / 4);
        xsG[tid] = target[bk * C_ + 2 * tid]     + (float)(H_ / 4);
        ysG[tid] = target[bk * C_ + 2 * tid + 1] + (float)(W_ / 4);
    }
    __syncthreads();

    // ---- deterministic tree reduce of |pred-target|*m over 256 threads ----
    for (int s = 128; s > 0; s >>= 1) {
        if (tid < s) redF[tid] += redF[tid + s];
        __syncthreads();
    }
    if (tid == 0) ws_reg[bk] = redF[0];

    // ---- rasterize + IoU counts ----
    const int waveId = tid >> 5;
    int accI = 0, accP = 0, accG = 0;

    // B matrix: all ones (layout-independent row-sum)
    v16h ones;
#pragma unroll
    for (int j = 0; j < 16; ++j) ones[j] = (_Float16)1.0f;

    // A layout (16-bit 16x32): lanes 0-15 -> row M=lane, halves 0..7 = K 0..7;
    // lanes 16-31 -> row M=lane-16, halves 0..7 = K 8..15. Halves 8..15 (K>=16) = 0.
    const int   ebase  = (lane < 16) ? 0 : 8;
    const float pxBase = (float)(lane & 15);

    for (int py = waveId; py < H_; py += 8) {
        const float pyf = (float)py;
        float xeP[8], xeG[8];
#pragma unroll
        for (int j = 0; j < 8; ++j) {
            const int e  = ebase + j;
            const int e2 = (e + 1) & (V_ - 1);
            {
                const float x1 = xsP[e], y1 = ysP[e], x2 = xsP[e2], y2 = ysP[e2];
                const bool crossing =
                    ((y1 <= pyf) & (y2 > pyf)) | ((y2 <= pyf) & (y1 > pyf));
                const float dy    = y2 - y1;
                const float denom = (dy == 0.0f) ? 1.0f : dy;
                const float slope = (x2 - x1) / denom;
                const float xint  = x1 + (pyf - y1) * slope;
                xeP[j] = crossing ? xint : -1.0f;   // px >= 0, so never < -1
            }
            {
                const float x1 = xsG[e], y1 = ysG[e], x2 = xsG[e2], y2 = ysG[e2];
                const bool crossing =
                    ((y1 <= pyf) & (y2 > pyf)) | ((y2 <= pyf) & (y1 > pyf));
                const float dy    = y2 - y1;
                const float denom = (dy == 0.0f) ? 1.0f : dy;
                const float slope = (x2 - x1) / denom;
                const float xint  = x1 + (pyf - y1) * slope;
                xeG[j] = crossing ? xint : -1.0f;
            }
        }

        for (int t = 0; t < 8; ++t) {            // 8 tiles of 16 px = 128 px row
            const float pxf = (float)(t * 16) + pxBase;
            v16h aP = {};
            v16h aG = {};
#pragma unroll
            for (int j = 0; j < 8; ++j) {
                aP[j] = (pxf < xeP[j]) ? (_Float16)1.0f : (_Float16)0.0f;
                aG[j] = (pxf < xeG[j]) ? (_Float16)1.0f : (_Float16)0.0f;
            }
            const v8f zero = {};
            // D[m][n] = sum_k A[m][k]  (edge-crossing count per pixel m)
            v8f dP = __builtin_amdgcn_wmma_f32_16x16x32_f16(
                false, aP, false, ones, (short)0, zero, false, false);
            v8f dG = __builtin_amdgcn_wmma_f32_16x16x32_f16(
                false, aG, false, ones, (short)0, zero, false, false);

            // f32 16x16 D layout: lane<16: VGPR r -> M=r; lane>=16: M=8+r.
            // Predicate so each of the 16 pixels is accumulated exactly once:
            // active lane for VGPR r is lane==r (pixel r) and lane==16+r (pixel 8+r).
#pragma unroll
            for (int r = 0; r < 8; ++r) {
                const int act = ((lane & 15) == r) ? 1 : 0;
                const int ip  = ((int)dP[r]) & 1;   // parity -> inside
                const int ig  = ((int)dG[r]) & 1;
                accI += act & ip & ig;
                accP += act & ip;
                accG += act & ig;
            }
        }
    }

    // ---- deterministic block reduce of counts ----
    redI[tid]       = accI;
    redI[256 + tid] = accP;
    redI[512 + tid] = accG;
    __syncthreads();
    for (int s = 128; s > 0; s >>= 1) {
        if (tid < s) {
            redI[tid]       += redI[tid + s];
            redI[256 + tid] += redI[256 + tid + s];
            redI[512 + tid] += redI[512 + tid + s];
        }
        __syncthreads();
    }
    if (tid == 0) {
        const float inter = (float)redI[0];
        const float uni   = (float)(redI[256] + redI[512]) - inter;
        const float iou   = inter / (uni + 1e-6f);
        ws_iou[bk] = iou * m;
    }
}

__global__ void __launch_bounds__(256) poly_finalize_kernel(
    const int*   __restrict__ mask,
    const float* __restrict__ ws_iou,
    const float* __restrict__ ws_reg,
    float*       __restrict__ out)
{
    __shared__ float sI[256], sR[256], sM[256];
    const int t = threadIdx.x;
    sI[t] = ws_iou[t] + ws_iou[t + 256];
    sR[t] = ws_reg[t] + ws_reg[t + 256];
    sM[t] = (float)mask[t] + (float)mask[t + 256];
    __syncthreads();
    for (int s = 128; s > 0; s >>= 1) {
        if (t < s) {
            sI[t] += sI[t + s];
            sR[t] += sR[t + s];
            sM[t] += sM[t + s];
        }
        __syncthreads();
    }
    if (t == 0) {
        const float msum = sM[0];
        const float loss     = 1.0f - sI[0] / (msum + 1e-6f);
        const float loss_reg = sR[0] / (msum + 1e-6f);
        out[0] = loss + loss_reg;
    }
}

extern "C" void kernel_launch(void* const* d_in, const int* in_sizes, int n_in,
                              void* d_out, int out_size, void* d_ws, size_t ws_size,
                              hipStream_t stream) {
    const float* output = (const float*)d_in[0];   // (B,C,H,W) f32
    const int*   mask   = (const int*)d_in[1];     // (B,K) i32
    const int*   ind    = (const int*)d_in[2];     // (B,K) i32
    const float* target = (const float*)d_in[3];   // (B,K,C) f32
    // d_in[4] = freq_mask, unused by the reference.

    float* ws_iou = (float*)d_ws;          // NPOLY floats
    float* ws_reg = ws_iou + NPOLY;        // NPOLY floats

    poly_iou_kernel<<<NPOLY, 256, 0, stream>>>(output, mask, ind, target,
                                               ws_iou, ws_reg);
    poly_finalize_kernel<<<1, 256, 0, stream>>>(mask, ws_iou, ws_reg,
                                                (float*)d_out);
}
